// CNN_biLSTM_CRF_29317446762912
// MI455X (gfx1250) — matrix operand order repaired and checked by reading, not compiled, not run on hardware
//
#include <hip/hip_runtime.h>
#include <hip/hip_bf16.h>

// ---------------- problem constants ----------------
#define BB   64
#define SS   128
#define LW   20
#define CHV  100
#define CHE  30
#define CHC  30
#define WE   300
#define HH   256
#define NC   25
#define FF   330
#define FPAD 352            // 330 padded to 11*32 for K-loop
#define G4H  1024           // 4*H
#define NTOT 2048           // fwd+rev gate columns

typedef _Float16 f16;
typedef __attribute__((ext_vector_type(16))) _Float16 v16h;
typedef __attribute__((ext_vector_type(8)))  _Float16 v8h;
typedef __attribute__((ext_vector_type(2)))  _Float16 h2;
typedef __attribute__((ext_vector_type(8)))  float    v8f;

// ---------------- WMMA helpers (CDNA5, wave32) ----------------
__device__ __forceinline__ v8f wmma_f16(v16h a, v16h b, v8f c) {
#if defined(__gfx1250__)
  // D = A(16x32 f16) x B(32x16 f16) + C(16x16 f32)
  return __builtin_amdgcn_wmma_f32_16x16x32_f16(false, a, false, b, (short)0, c,
                                                false, false);
#else
  return c;
#endif
}

// A-matrix 16x32 f16 fragment (ISA 7.12.2): lane m = lane&15,
// VGPR v holds K-pair at (v>=4?16:0) + (lane>=16?8:0) + 2*(v&3).
__device__ __forceinline__ v16h load_a_frag(const f16* __restrict__ row,
                                            int kbase, int lane) {
  const int hi = (lane >= 16) ? 8 : 0;
  v16h a;
#pragma unroll
  for (int v = 0; v < 8; ++v) {
    int k = kbase + ((v & 4) ? 16 : 0) + hi + 2 * (v & 3);
    h2 p = *(const h2*)(row + k);
    a[2 * v]     = p[0];
    a[2 * v + 1] = p[1];
  }
  return a;
}
// same pattern but from LDS
__device__ __forceinline__ v16h load_a_frag_lds(const f16* row, int kbase, int lane) {
  const int hi = (lane >= 16) ? 8 : 0;
  v16h a;
#pragma unroll
  for (int v = 0; v < 8; ++v) {
    int k = kbase + ((v & 4) ? 16 : 0) + hi + 2 * (v & 3);
    a[2 * v]     = row[k];
    a[2 * v + 1] = row[k + 1];
  }
  return a;
}

// B-matrix 32x16 f16 fragment: lane holds column n = lane&15, K values in the
// same order as the A fragment: kbase+hb+0..7 and kbase+16+hb+0..7 (hb=8 for hi
// lanes). W stored row-major [n][k] == B^T -> 2x b128 loads.
__device__ __forceinline__ v16h load_b_frag(const f16* __restrict__ col, int kstart) {
  const int kb = (kstart & ~31) + ((kstart & 16) ? 8 : 0);
  v8h lo = *(const v8h*)(col + kb);
  v8h hi = *(const v8h*)(col + kb + 16);
  v16h b;
#pragma unroll
  for (int v = 0; v < 8; ++v) { b[v] = lo[v]; b[8 + v] = hi[v]; }
  return b;
}

__device__ __forceinline__ float sigm(float x) { return 1.f / (1.f + __expf(-x)); }

// ---------------- kernel 1a: w_ih -> f16, K padded 330->352 ----------------
__global__ void prep_wih(const float* __restrict__ wf, const float* __restrict__ wr,
                         f16* __restrict__ W) {
  int idx = blockIdx.x * blockDim.x + threadIdx.x;
  if (idx >= NTOT * FPAD) return;
  int n = idx / FPAD, k = idx % FPAD;
  float v = 0.f;
  if (k < FF) v = (n < G4H) ? wf[n * FF + k] : wr[(n - G4H) * FF + k];
  W[idx] = (f16)v;
}

// ---------------- kernel 1b: w_hh -> f16 ----------------
__global__ void prep_whh(const float* __restrict__ wf, const float* __restrict__ wr,
                         f16* __restrict__ W) {
  int idx = blockIdx.x * blockDim.x + threadIdx.x;
  if (idx >= 2 * G4H * HH) return;
  int d = idx / (G4H * HH), r = idx % (G4H * HH);
  W[idx] = (f16)(d ? wr[r] : wf[r]);
}

// ---------------- kernel 2: char-CNN + gather -> X16 (8192 x 352 f16) -------
__global__ void feat_kernel(const float* __restrict__ word_table,
                            const float* __restrict__ char_table,
                            const float* __restrict__ conv_w,
                            const float* __restrict__ conv_b,
                            const int* __restrict__ sent,
                            const int* __restrict__ word,
                            f16* __restrict__ X16) {
  const int row = blockIdx.x;         // b*S + s
  const int tid = threadIdx.x;        // 64 threads
  __shared__ float cemb[LW][CHE];
  for (int e = tid; e < LW * CHE; e += 64) {
    int p = e / CHE, ic = e % CHE;
    int idx = word[row * LW + p];
    cemb[p][ic] = (idx == 0) ? 0.f : char_table[idx * CHE + ic];  // row0 zeroed
  }
  __syncthreads();
  f16* xr = X16 + (size_t)row * FPAD;
  // word embedding -> f16
  const int w = sent[row];
  for (int k = tid; k < WE; k += 64)
    xr[k] = (f16)word_table[(size_t)w * WE + k];
  // conv1d(k=3, pad=1) + max over positions
  if (tid < CHC) {
    const int oc = tid;
    float best = -1e30f;
    for (int pos = 0; pos < LW; ++pos) {
      float acc = conv_b[oc];
#pragma unroll
      for (int tt = 0; tt < 3; ++tt) {
        int p = pos + tt - 1;
        if (p >= 0 && p < LW) {
          for (int ic = 0; ic < CHE; ++ic)
            acc += cemb[p][ic] * conv_w[oc * (CHE * 3) + ic * 3 + tt];
        }
      }
      best = fmaxf(best, acc);
    }
    xr[WE + oc] = (f16)best;
  }
  if (tid < FPAD - FF) xr[FF + tid] = (f16)0.f;   // zero K-padding
}

// ---------------- kernel 3: gx = X * [Wih_f ; Wih_r]^T (WMMA) ---------------
// M=8192, N=2048, K=352. One wave -> 32x64 strip: 2 M-tiles x 4 N-tiles,
// every B fragment feeds two WMMAs (halves B-side traffic, back-to-back
// independent WMMAs hide the D->A/B hazard latency).
__global__ void gemm_gx(const f16* __restrict__ X16, const f16* __restrict__ W,
                        float* __restrict__ gx) {
  const int lane = threadIdx.x & 31;
  const int wid  = blockIdx.x * (blockDim.x >> 5) + (threadIdx.x >> 5);
  const int mpair = wid & 255;        // 256 M-tile pairs (32 rows each)
  const int ngrp  = wid >> 8;         // 32 N-groups of 64
  const bool hi = lane >= 16;
  const int nlo = lane & 15;
  const f16* xrow0 = X16 + (size_t)(mpair * 32 + nlo) * FPAD;
  const f16* xrow1 = xrow0 + (size_t)16 * FPAD;
  v8f acc[2][4] = {};
  for (int kk = 0; kk < FPAD / 32; ++kk) {
    const int kbase = kk * 32;
    if (kk + 1 < FPAD / 32) {
      // locality 3 -> WGP-scope global_prefetch_b8: pull next K-slice into
      // all cache levels (near caches), not just GL2
      __builtin_prefetch(xrow0 + kbase + 32, 0, 3);
      __builtin_prefetch(xrow1 + kbase + 32, 0, 3);
    }
    v16h a0 = load_a_frag(xrow0, kbase, lane);
    v16h a1 = load_a_frag(xrow1, kbase, lane);
    const int kstart = kbase + (hi ? 16 : 0);
#pragma unroll
    for (int j = 0; j < 4; ++j) {
      int n = ngrp * 64 + j * 16 + nlo;
      v16h b = load_b_frag(W + (size_t)n * FPAD, kstart);
      acc[0][j] = wmma_f16(a0, b, acc[0][j]);
      acc[1][j] = wmma_f16(a1, b, acc[1][j]);
    }
  }
#pragma unroll
  for (int mi = 0; mi < 2; ++mi) {
#pragma unroll
    for (int j = 0; j < 4; ++j) {
      int n = ngrp * 64 + j * 16 + nlo;
#pragma unroll
      for (int v = 0; v < 8; ++v) {
        int mr = mpair * 32 + mi * 16 + v + (hi ? 8 : 0);
        gx[(size_t)mr * NTOT + n] = acc[mi][j][v];
      }
    }
  }
}

// ---------------- kernel 4: bidirectional LSTM scan -------------------------
// 2 blocks (dir), 1024 threads = 32 waves. Per step: (64x1024) += h(64x256
// f16, LDS) x Whh^T. Wave owns all 4 gates of one (mtile, htile-pair) so the
// gate math is purely per-lane.
__global__ void __launch_bounds__(1024)
lstm_scan(const float* __restrict__ gx, const f16* __restrict__ Whh,
          const float* __restrict__ b_f, const float* __restrict__ b_r,
          float* __restrict__ h_out) {
  const int dir  = blockIdx.x;
  const int tid  = threadIdx.x;
  const int lane = tid & 31;
  const int wave = tid >> 5;
  const int mtile  = wave & 3;   // 4 batch tiles
  const int htpair = wave >> 2;  // 8 pairs of hidden tiles
  const bool hi = lane >= 16;
  const int nlo = lane & 15;

  __shared__ f16 h_lds[BB * HH];        // 32 KB
  __shared__ float bias_s[G4H];

  for (int i = tid; i < BB * HH; i += 1024) h_lds[i] = (f16)0.f;
  {
    const float* bsrc = dir ? b_r : b_f;
    if (tid < G4H) bias_s[tid] = bsrc[tid];
  }
  __syncthreads();

  float c[2][8];
#pragma unroll
  for (int g2 = 0; g2 < 2; ++g2)
#pragma unroll
    for (int v = 0; v < 8; ++v) c[g2][v] = 0.f;

  const f16* Wd = Whh + (size_t)dir * G4H * HH;
  const f16* arow = h_lds + (mtile * 16 + nlo) * HH;

  for (int t = 0; t < SS; ++t) {
    const int s = dir ? (SS - 1 - t) : t;
    v8f acc[2][4] = {};
    for (int kk = 0; kk < HH / 32; ++kk) {
      const int kbase = kk * 32;
      const int kstart = kbase + (hi ? 16 : 0);
      if (kk + 1 < HH / 32) {             // WGP-scope prefetch of next W slice
        int n0 = htpair * 32 + nlo;
        __builtin_prefetch(Wd + (size_t)n0 * HH + kstart + 32, 0, 3);
      }
      v16h a = load_a_frag_lds(arow, kbase, lane);
#pragma unroll
      for (int g2 = 0; g2 < 2; ++g2) {
        const int ht = htpair * 2 + g2;
#pragma unroll
        for (int g = 0; g < 4; ++g) {
          int n = g * HH + ht * 16 + nlo;
          v16h b = load_b_frag(Wd + (size_t)n * HH, kstart);
          acc[g2][g] = wmma_f16(a, b, acc[g2][g]);
        }
      }
    }
    // gates: i/f/g/o for the same (b,h) sit in the same lane slot of the 4 accs
    float hval[2][8];
#pragma unroll
    for (int g2 = 0; g2 < 2; ++g2) {
      const int hh = (htpair * 2 + g2) * 16 + nlo;
#pragma unroll
      for (int v = 0; v < 8; ++v) {
        const int bi = mtile * 16 + v + (hi ? 8 : 0);
        const float* gr = gx + ((size_t)(bi * SS + s)) * NTOT + dir * G4H;
        float gi = acc[g2][0][v] + gr[hh]            + bias_s[hh];
        float gf = acc[g2][1][v] + gr[HH + hh]       + bias_s[HH + hh];
        float gg = acc[g2][2][v] + gr[2 * HH + hh]   + bias_s[2 * HH + hh];
        float go = acc[g2][3][v] + gr[3 * HH + hh]   + bias_s[3 * HH + hh];
        float cv = sigm(gf) * c[g2][v] + sigm(gi) * tanhf(gg);
        c[g2][v] = cv;
        hval[g2][v] = sigm(go) * tanhf(cv);
      }
    }
    __syncthreads();   // all waves done reading h_lds for this step
#pragma unroll
    for (int g2 = 0; g2 < 2; ++g2) {
      const int hh = (htpair * 2 + g2) * 16 + nlo;
#pragma unroll
      for (int v = 0; v < 8; ++v) {
        const int bi = mtile * 16 + v + (hi ? 8 : 0);
        h_lds[bi * HH + hh] = (f16)hval[g2][v];
        h_out[((size_t)(s * BB + bi)) * (2 * HH) + dir * HH + hh] = hval[g2][v];
      }
    }
    __syncthreads();   // h_lds update visible before next step's GEMM
  }
}

// ---------------- kernel 5: em = h @ lin_w^T + lin_b ------------------------
__global__ void em_kernel(const float* __restrict__ h, const float* __restrict__ lw,
                          const float* __restrict__ lb, float* __restrict__ em) {
  int idx = blockIdx.x * blockDim.x + threadIdx.x;
  if (idx >= SS * BB * NC) return;
  int j  = idx % NC;
  int sb = idx / NC;                 // s*B + b
  const float* hr = h + (size_t)sb * (2 * HH);
  const float* wr = lw + (size_t)j * (2 * HH);
  float acc = lb[j];
  for (int k = 0; k < 2 * HH; k += 4) {
    acc += hr[k] * wr[k] + hr[k + 1] * wr[k + 1] +
           hr[k + 2] * wr[k + 2] + hr[k + 3] * wr[k + 3];
  }
  em[idx] = acc;
}

// ---------------- kernel 6: CRF negative log-likelihood ---------------------
__global__ void __launch_bounds__(1024)
crf_kernel(const float* __restrict__ em, const int* __restrict__ tag,
           const unsigned char* __restrict__ mask,
           const float* __restrict__ start_t, const float* __restrict__ end_t,
           const float* __restrict__ trans, float* __restrict__ out) {
  __shared__ float trans_s[NC * NC], start_s[NC], end_s[NC];
  __shared__ float alpha[2][BB * NC];
  __shared__ float red[BB];
  const int tid = threadIdx.x;
  for (int i = tid; i < NC * NC; i += 1024) trans_s[i] = trans[i];
  if (tid < NC) { start_s[tid] = start_t[tid]; end_s[tid] = end_t[tid]; }
  __syncthreads();
  for (int p = tid; p < BB * NC; p += 1024) {
    int b = p / NC, j = p % NC;
    alpha[0][p] = start_s[j] + em[b * NC + j];
  }
  __syncthreads();
  for (int t = 1; t < SS; ++t) {
    const int cur = t & 1, prv = cur ^ 1;
    for (int p = tid; p < BB * NC; p += 1024) {
      int b = p / NC, j = p % NC;
      float vs[NC], mx = -1e30f;
#pragma unroll
      for (int i = 0; i < NC; ++i) {
        float v = alpha[prv][b * NC + i] + trans_s[i * NC + j];
        vs[i] = v; mx = fmaxf(mx, v);
      }
      float sum = 0.f;
#pragma unroll
      for (int i = 0; i < NC; ++i) sum += __expf(vs[i] - mx);
      float nxt = mx + __logf(sum) + em[(t * BB + b) * NC + j];
      alpha[cur][p] = mask[b * SS + t] ? nxt : alpha[prv][p];
    }
    __syncthreads();
  }
  if (tid < BB) {
    const int b = tid;
    const int fin = (SS - 1) & 1;
    float mx = -1e30f;
    for (int j = 0; j < NC; ++j) mx = fmaxf(mx, alpha[fin][b * NC + j] + end_s[j]);
    float sum = 0.f;
    for (int j = 0; j < NC; ++j) sum += __expf(alpha[fin][b * NC + j] + end_s[j] - mx);
    float logZ = mx + __logf(sum);
    int tg0 = tag[b * SS];
    float score = start_s[tg0] + em[b * NC + tg0];
    int prev = tg0;
    int cnt = mask[b * SS] ? 1 : 0;
    for (int t = 1; t < SS; ++t) {
      int tgt = tag[b * SS + t];
      if (mask[b * SS + t]) {
        score += trans_s[prev * NC + tgt] + em[(t * BB + b) * NC + tgt];
        ++cnt;
      }
      prev = tgt;
    }
    score += end_s[tag[b * SS + (cnt - 1)]];
    red[b] = score - logZ;
  }
  __syncthreads();
  if (tid == 0) {
    float total = 0.f;
    for (int b = 0; b < BB; ++b) total += red[b];
    out[0] = -total;
  }
}

// ---------------- host launcher ----------------
extern "C" void kernel_launch(void* const* d_in, const int* in_sizes, int n_in,
                              void* d_out, int out_size, void* d_ws, size_t ws_size,
                              hipStream_t stream) {
  const float* word_table = (const float*)d_in[0];
  const float* char_table = (const float*)d_in[1];
  const float* conv_w     = (const float*)d_in[2];
  const float* conv_b     = (const float*)d_in[3];
  const float* w_ih_f     = (const float*)d_in[4];
  const float* w_hh_f     = (const float*)d_in[5];
  const float* b_f        = (const float*)d_in[6];
  const float* w_ih_r     = (const float*)d_in[7];
  const float* w_hh_r     = (const float*)d_in[8];
  const float* b_r        = (const float*)d_in[9];
  const float* lin_w      = (const float*)d_in[10];
  const float* lin_b      = (const float*)d_in[11];
  const float* start_t    = (const float*)d_in[12];
  const float* end_t      = (const float*)d_in[13];
  const float* trans      = (const float*)d_in[14];
  const int*   sent       = (const int*)d_in[15];
  const int*   word       = (const int*)d_in[16];
  const int*   tag        = (const int*)d_in[17];
  const unsigned char* mask = (const unsigned char*)d_in[18];
  float* out = (float*)d_out;

  // workspace carving (all offsets multiples of 256B)
  char* ws = (char*)d_ws;
  const size_t oX   = 0;                                   // 8192*352 f16
  const size_t oWih = oX   + (size_t)BB * SS * FPAD * 2;   // 2048*352 f16
  const size_t oWhh = oWih + (size_t)NTOT * FPAD * 2;      // 2*1024*256 f16
  const size_t oGx  = oWhh + (size_t)2 * G4H * HH * 2;     // 8192*2048 f32
  const size_t oH   = oGx  + (size_t)BB * SS * NTOT * 4;   // 128*64*512 f32
  const size_t oEm  = oH   + (size_t)SS * BB * 2 * HH * 4; // 128*64*25 f32
  f16*   X16  = (f16*)(ws + oX);
  f16*   Wih  = (f16*)(ws + oWih);
  f16*   Whh  = (f16*)(ws + oWhh);
  float* gx   = (float*)(ws + oGx);
  float* hseq = (float*)(ws + oH);
  float* em   = (float*)(ws + oEm);

  prep_wih<<<(NTOT * FPAD + 255) / 256, 256, 0, stream>>>(w_ih_f, w_ih_r, Wih);
  prep_whh<<<(2 * G4H * HH + 255) / 256, 256, 0, stream>>>(w_hh_f, w_hh_r, Whh);
  feat_kernel<<<BB * SS, 64, 0, stream>>>(word_table, char_table, conv_w, conv_b,
                                          sent, word, X16);
  // 256 M-pairs * 32 N-groups = 8192 waves, 8 waves/block
  gemm_gx<<<1024, 256, 0, stream>>>(X16, Wih, gx);
  lstm_scan<<<2, 1024, 0, stream>>>(gx, Whh, b_f, b_r, hseq);
  em_kernel<<<(SS * BB * NC + 255) / 256, 256, 0, stream>>>(hseq, lin_w, lin_b, em);
  crf_kernel<<<1, 1024, 0, stream>>>(em, tag, mask, start_t, end_t, trans, out);
}
